// NNhead_19164144075037
// MI455X (gfx1250) — matrix operand order repaired and checked by reading, not compiled, not run on hardware
//
#include <hip/hip_runtime.h>

#define DDIM 1024
#define SBUF 256
#define KC   64           // k-chunk per stage = 2 x bf16 WMMA depth
#define BM   64           // rows per block
#define LDA  72           // padded LDS row stride in ushorts (144B = 9 quads, conflict-free)

typedef __attribute__((ext_vector_type(16))) __bf16 v16bf;
typedef __attribute__((ext_vector_type(8)))  float  v8f;
typedef __attribute__((ext_vector_type(8)))  unsigned short v8us;
typedef __attribute__((ext_vector_type(4)))  float  f32x4;
typedef __attribute__((ext_vector_type(4)))  __bf16 bf16x4;

__device__ __forceinline__ v16bf ld_frag(const unsigned short* p0, const unsigned short* p1) {
    union { v16bf bf; v8us u[2]; } u;
    u.u[0] = *(const v8us*)p0;   // ds_load_b128
    u.u[1] = *(const v8us*)p1;   // ds_load_b128
    return u.bf;
}

// Per row: inv[r] = 1/||row|| and (optionally) bf16 copy of the raw row into dst16.
__global__ __launch_bounds__(256) void prep_kernel(const float* __restrict__ src,
                                                   unsigned short* __restrict__ dst16,
                                                   float* __restrict__ inv) {
    const int r   = blockIdx.x;
    const int tid = threadIdx.x;
    const f32x4 v = *(const f32x4*)(src + (size_t)r * DDIM + tid * 4);
    if (dst16) {   // uniform branch
        bf16x4 b = __builtin_convertvector(v, bf16x4);   // v_cvt_pk_bf16_f32
        *(bf16x4*)(dst16 + (size_t)r * DDIM + tid * 4) = b;
    }
    float s = v.x * v.x;
    s = fmaf(v.y, v.y, s);
    s = fmaf(v.z, v.z, s);
    s = fmaf(v.w, v.w, s);
    for (int m = 16; m >= 1; m >>= 1) s += __shfl_xor(s, m, 32);
    __shared__ float red[8];
    if ((tid & 31) == 0) red[tid >> 5] = s;
    __syncthreads();
    if (tid == 0) {
        float t = 0.f;
        for (int i = 0; i < 8; ++i) t += red[i];
        inv[r] = rsqrtf(t);
    }
}

// One block: 64 input rows x one task's 256 prototypes. 8 waves (wave32).
// PREPACKED: double-buffered async global->LDS staging overlapped with WMMA.
template <bool PREPACKED>
__global__ __launch_bounds__(256) void nnhead_wmma_kernel(
    const float* __restrict__ Xf, const float* __restrict__ Wf,
    const unsigned short* __restrict__ Xh, const unsigned short* __restrict__ Wh,
    const float* __restrict__ invx, const float* __restrict__ invw,
    float* __restrict__ out, int T)
{
    __shared__ __align__(16) unsigned short Abuf[2][BM   * LDA];   // 2 x  9 KB
    __shared__ __align__(16) unsigned short Bbuf[2][SBUF * LDA];   // 2 x 36 KB
    __shared__ float wavemin[8][16];

    const int tid  = threadIdx.x;
    const int lane = tid & 31;
    const int w    = tid >> 5;
    const int half = lane >> 4;
    const int ln   = lane & 15;

    const int m0 = blockIdx.x * BM;
    const int t  = blockIdx.y;

    const v8f zero = {0.f, 0.f, 0.f, 0.f, 0.f, 0.f, 0.f, 0.f};
    v8f acc[8];
#pragma unroll
    for (int j = 0; j < 8; ++j) acc[j] = zero;

    const int aRow  = (w & 3) * 16 + ln;   // A-fragment row (m) for this lane
    const int nBase = (w >> 2) * 128;      // this wave's column base within the task

    const unsigned short* Xbase = Xh + (size_t)m0 * DDIM;
    const unsigned short* Wbase = Wh + ((size_t)t * SBUF) * DDIM;

    // issue 10 async 16B copies (2 A-chunks + 8 B-chunks) for chunk k0 into buffer p
    auto issue_stage = [&](int p, int k0) {
        const unsigned ldsA = (unsigned)(unsigned long long)&Abuf[p][0];
        const unsigned ldsB = (unsigned)(unsigned long long)&Bbuf[p][0];
#pragma unroll
        for (int it = 0; it < 2; ++it) {           // A: 64 rows x 64 k = 512 x 16B
            int i = tid + it * 256, row = i >> 3, c = i & 7;
            const unsigned short* g = Xbase + (size_t)row * DDIM + k0 + c * 8;
            unsigned lo = ldsA + (unsigned)(row * LDA + c * 8) * 2;
            asm volatile("global_load_async_to_lds_b128 %0, %1, off"
                         :: "v"(lo), "v"(g) : "memory");
        }
#pragma unroll
        for (int it = 0; it < 8; ++it) {           // B: 256 rows x 64 k = 2048 x 16B
            int i = tid + it * 256, row = i >> 3, c = i & 7;
            const unsigned short* g = Wbase + (size_t)row * DDIM + k0 + c * 8;
            unsigned lo = ldsB + (unsigned)(row * LDA + c * 8) * 2;
            asm volatile("global_load_async_to_lds_b128 %0, %1, off"
                         :: "v"(lo), "v"(g) : "memory");
        }
    };

    const int KSTEPS = DDIM / KC;

    if constexpr (PREPACKED) issue_stage(0, 0);

    for (int ks = 0; ks < KSTEPS; ++ks) {
        const int p = ks & 1;

        if constexpr (PREPACKED) {
            if (ks + 1 < KSTEPS) {
                issue_stage(p ^ 1, (ks + 1) * KC);   // prefetch next chunk into other buffer
                // drain only the PREVIOUS 10 issues (in-order completion): current buffer ready
                asm volatile("s_wait_asynccnt 10" ::: "memory");
            } else {
                asm volatile("s_wait_asynccnt 0" ::: "memory");
            }
        } else {
            // ---- fallback: load f32, v_cvt_pk_bf16_f32, store to LDS (buffer 0) ----
            const int k0 = ks * KC;
            const float* Xp = Xf + (size_t)m0 * DDIM + k0;
            const float* Wp = Wf + ((size_t)t * SBUF) * DDIM + k0;
#pragma unroll
            for (int it = 0; it < 4; ++it) {       // A: 64x64 f32 = 1024 float4
                int i = tid + it * 256, row = i >> 4, c4 = i & 15;
                f32x4 f = *(const f32x4*)(Xp + (size_t)row * DDIM + c4 * 4);
                *(bf16x4*)&Abuf[0][row * LDA + c4 * 4] = __builtin_convertvector(f, bf16x4);
            }
#pragma unroll
            for (int it = 0; it < 16; ++it) {      // B: 256x64 f32 = 4096 float4
                int i = tid + it * 256, row = i >> 4, c4 = i & 15;
                f32x4 f = *(const f32x4*)(Wp + (size_t)row * DDIM + c4 * 4);
                *(bf16x4*)&Bbuf[0][row * LDA + c4 * 4] = __builtin_convertvector(f, bf16x4);
            }
            if (ks + 1 < KSTEPS) {   // global_prefetch_b8 of next k-chunk
                __builtin_prefetch(Xp + (size_t)(tid >> 2) * DDIM + KC + (tid & 3) * 16, 0, 1);
                __builtin_prefetch(Wp + (size_t)tid * DDIM + KC, 0, 1);
            }
        }
        __syncthreads();

        const unsigned short* Ab = PREPACKED ? Abuf[p] : Abuf[0];
        const unsigned short* Bb = PREPACKED ? Bbuf[p] : Bbuf[0];
#pragma unroll
        for (int kk = 0; kk < 2; ++kk) {            // two WMMA depths per staged chunk
            // A frag (16x32 bf16, ISA 7.12.2): lane(m) chunks at k = 8*half, 16+8*half
            const int abase = aRow * LDA + kk * 32;
            const v16bf a = ld_frag(&Ab[abase + 8 * half],
                                    &Ab[abase + 16 + 8 * half]);
#pragma unroll
            for (int j = 0; j < 8; ++j) {
                // B frag (32x16 bf16): lane(n) chunks at k = 8*half, 16+8*half (matches A)
                const int n = nBase + j * 16 + ln;
                const int bbase = n * LDA + kk * 32 + 8 * half;
                const v16bf b = ld_frag(&Bb[bbase], &Bb[bbase + 16]);
                acc[j] = __builtin_amdgcn_wmma_f32_16x16x32_bf16(
                    false, a, false, b, (short)0, acc[j], false, false);
            }
        }
        __syncthreads();   // all waves done reading buffer p before it is re-filled
    }

    // ---- epilogue: dist = sqrt(max(2 - 2*dot_hat, eps)); min over n; out = -min ----
    const int mStrip = m0 + (w & 3) * 16;
    float invwv[8];
#pragma unroll
    for (int j = 0; j < 8; ++j) invwv[j] = invw[t * SBUF + nBase + j * 16 + ln];

    float rmin[8];
#pragma unroll
    for (int r = 0; r < 8; ++r) {
        const int gm = mStrip + r + 8 * half;   // C/D layout: VGPR r -> M = r + 8*half
        const float ix = invx[gm];
        float mn = 3.4e38f;
#pragma unroll
        for (int j = 0; j < 8; ++j) {
            float dot = acc[j][r] * ix * invwv[j];
            float sq  = fmaxf(2.f - 2.f * dot, 1e-12f);
            mn = fminf(mn, sqrtf(sq));
        }
#pragma unroll
        for (int msk = 8; msk >= 1; msk >>= 1)
            mn = fminf(mn, __shfl_xor(mn, msk, 32));
        rmin[r] = mn;
    }
    if (ln == 0) {
#pragma unroll
        for (int r = 0; r < 8; ++r) wavemin[w][r + 8 * half] = rmin[r];
    }
    __syncthreads();
    if (tid < 64) {
        const int s = tid >> 4;
        const float v = fminf(wavemin[s][tid & 15], wavemin[s + 4][tid & 15]);
        out[(size_t)(m0 + tid) * T + t] = -v;
    }
}

extern "C" void kernel_launch(void* const* d_in, const int* in_sizes, int n_in,
                              void* d_out, int out_size, void* d_ws, size_t ws_size,
                              hipStream_t stream) {
    const float* X = (const float*)d_in[0];   // (B, 1024) f32
    const float* W = (const float*)d_in[1];   // (T*256, 1024) f32
    const int Brows = in_sizes[0] / DDIM;     // 8192
    const int WRows = in_sizes[1] / DDIM;     // 5120
    const int T     = WRows / SBUF;           // 20

    float* invx = (float*)d_ws;               // Brows floats
    float* invw = invx + Brows;               // WRows floats

    const size_t normBytes = (size_t)(Brows + WRows) * sizeof(float);
    const size_t packOff   = (normBytes + 255) & ~(size_t)255;
    const size_t packBytes = packOff + (size_t)(Brows + WRows) * DDIM * 2;
    const bool   prepack   = ws_size >= packBytes;

    unsigned short* X16 = prepack ? (unsigned short*)((char*)d_ws + packOff) : nullptr;
    unsigned short* W16 = prepack ? X16 + (size_t)Brows * DDIM : nullptr;

    prep_kernel<<<Brows, 256, 0, stream>>>(X, X16, invx);
    prep_kernel<<<WRows, 256, 0, stream>>>(W, W16, invw);

    dim3 grid(Brows / BM, T);
    if (prepack) {
        nnhead_wmma_kernel<true><<<grid, 256, 0, stream>>>(
            X, W, X16, W16, invx, invw, (float*)d_out, T);
    } else {
        nnhead_wmma_kernel<false><<<grid, 256, 0, stream>>>(
            X, W, X16, W16, invx, invw, (float*)d_out, T);
    }
}
